// IterativeDecomposer_1597727834419
// MI455X (gfx1250) — matrix operand order repaired and checked by reading, not compiled, not run on hardware
//
#include <hip/hip_runtime.h>
#include <hip/hip_bf16.h>
#include <math.h>

typedef __attribute__((ext_vector_type(16))) _Float16 v16h;
typedef __attribute__((ext_vector_type(8)))  float    v8f;
typedef _Float16 half_t;

#define N_SAMPLES 32768
#define HOP       256
#define WIN       512
#define TFR       128     // T_FRAMES
#define CCH       1024    // spec channels
#define NEV       8
#define NATOM     4096
#define WAVE_LEN  8192
#define EMB_D     256

union H16 { v16h v; uint4 u[2]; };
union HP2 { uint  u; half_t h[2]; };
union HP4 { uint2 u; half_t h[4]; };

// ---------------------------------------------------------------------------
// Weight conversion / packing
// ---------------------------------------------------------------------------
__global__ void cvt_f16_kernel(const float* __restrict__ src, half_t* __restrict__ dst, int nel) {
    for (int i = blockIdx.x * blockDim.x + threadIdx.x; i < nel; i += gridDim.x * blockDim.x)
        dst[i] = (half_t)src[i];
}

// enc_w: [8][1024][1024][2]  ->  ewh: [8][2][1024][1024] f16 (contiguous per side)
__global__ void pack_encw_kernel(const float* __restrict__ src, half_t* __restrict__ dst) {
    const size_t total = (size_t)8 * 1024 * 1024 * 2;
    for (size_t i = (size_t)blockIdx.x * blockDim.x + threadIdx.x; i < total;
         i += (size_t)gridDim.x * blockDim.x) {
        size_t c    = i & 1023;
        size_t o    = (i >> 10) & 1023;
        size_t side = (i >> 20) & 1;
        size_t l    = i >> 21;
        dst[i] = (half_t)src[(((l * 1024 + o) * 1024) + c) * 2 + side];
    }
}

// ---------------------------------------------------------------------------
// Fragment helpers (ISA 16-bit WMMA layouts, wave32)
//   A 16x32: lane m=l&15, halves 0..7 -> K=kbA..kbA+7, halves 8..15 -> K=16+kbA..
//            kbA = (l<16)?0:8
//   B 32x16: lane n=l&15, halves 0..7 -> K=kbB/2.., 8..15 -> K=16+kbB/2.., kbB = (l<16)?0:16
//   C 16x16: vgpr r -> M = r + ((l<16)?0:8), N = l&15
// LDS tiles: A as [row][40] (k-major), B as [t][40] (k-minor, transposed) so
// every fragment is two aligned ds_load_b128.
// ---------------------------------------------------------------------------
__device__ __forceinline__ v16h load_a_frag(const half_t* rowp, int kbA) {
    H16 t;
    t.u[0] = *(const uint4*)(rowp + kbA);
    t.u[1] = *(const uint4*)(rowp + 16 + kbA);
    return t.v;
}
__device__ __forceinline__ v16h load_b_frag(const half_t* colp, int kbB) {
    H16 t;
    t.u[0] = *(const uint4*)(colp + (kbB >> 1));
    t.u[1] = *(const uint4*)(colp + 16 + (kbB >> 1));
    return t.v;
}
__device__ __forceinline__ v16h zero_frag() {
    v16h z;
#pragma unroll
    for (int q = 0; q < 16; ++q) z[q] = (half_t)0.f;
    return z;
}

// ---------------------------------------------------------------------------
// transform: spec[b,c,t] (+/-)= | sum_f fb[c,f] * sig[b, t*HOP + f] |
// WMMA GEMM: M=1024 (c), K=512 (f), N=128 (t). Block: 128x128 tile, 8 waves.
// ---------------------------------------------------------------------------
__global__ __launch_bounds__(256) void transform_wmma_kernel(
    const float* __restrict__ sig,   // [2][32768]
    const half_t* __restrict__ fbh,  // [1024][512] f16
    float* __restrict__ spec,        // [2][1024][128]
    int sub)
{
    const int b  = blockIdx.y;
    const int m0 = blockIdx.x * 128;
    const float* sb = sig + (size_t)b * N_SAMPLES;

    __shared__ half_t As[128][40];   // [c_local][k]
    __shared__ half_t Bs[128][40];   // [t][k]  (transposed staging)

    const int tid  = threadIdx.x;
    const int lane = tid & 31;
    const int wv   = tid >> 5;
    const int wm   = wv & 1;
    const int wn   = wv >> 1;

    const int fm  = lane & 15;
    const int kbA = (lane < 16) ? 0 : 8;
    const int kbB = (lane < 16) ? 0 : 16;
    const int fn  = lane & 15;

    v8f vzero = {0.f, 0.f, 0.f, 0.f, 0.f, 0.f, 0.f, 0.f};
    v8f acc[4][2];
    for (int i = 0; i < 4; ++i) for (int j = 0; j < 2; ++j) acc[i][j] = vzero;

    const int ar0 = tid >> 2, ack = (tid & 3) * 8;  // A staging task

    for (int kk = 0; kk < WIN; kk += 32) {
        // Stage A (filterbank f16): grouped 16B copies, 2 rows/thread
        uint4 f0 = *(const uint4*)(fbh + (size_t)(m0 + ar0) * WIN + kk + ack);
        uint4 f1 = *(const uint4*)(fbh + (size_t)(m0 + ar0 + 64) * WIN + kk + ack);
        // Stage B (framed signal) transposed: consecutive f == consecutive samples
        float4 va[4];
        int tt[4], rr[4];
#pragma unroll
        for (int q = 0; q < 4; ++q) {
            int i = tid + 256 * q;
            tt[q] = i >> 3;
            rr[q] = (i & 7) * 4;
            int s0 = tt[q] * HOP + kk + rr[q];
            if (s0 + 3 < N_SAMPLES) {
                va[q] = *(const float4*)(sb + s0);
            } else {
                float4 w;
                w.x = (s0 + 0 < N_SAMPLES) ? sb[s0 + 0] : 0.f;
                w.y = (s0 + 1 < N_SAMPLES) ? sb[s0 + 1] : 0.f;
                w.z = (s0 + 2 < N_SAMPLES) ? sb[s0 + 2] : 0.f;
                w.w = (s0 + 3 < N_SAMPLES) ? sb[s0 + 3] : 0.f;
                va[q] = w;
            }
        }
        *(uint4*)&As[ar0][ack]      = f0;
        *(uint4*)&As[ar0 + 64][ack] = f1;
#pragma unroll
        for (int q = 0; q < 4; ++q) {
            HP4 p;
            p.h[0] = (half_t)va[q].x; p.h[1] = (half_t)va[q].y;
            p.h[2] = (half_t)va[q].z; p.h[3] = (half_t)va[q].w;
            *(uint2*)&Bs[tt[q]][rr[q]] = p.u;
        }
        __syncthreads();

        v16h bf[2];
#pragma unroll
        for (int j = 0; j < 2; ++j)
            bf[j] = load_b_frag(&Bs[32 * wn + 16 * j + fn][0], kbB);
#pragma unroll
        for (int i = 0; i < 4; ++i) {
            v16h af = load_a_frag(&As[64 * wm + 16 * i + fm][0], kbA);
#pragma unroll
            for (int j = 0; j < 2; ++j)
                acc[i][j] = __builtin_amdgcn_wmma_f32_16x16x32_f16(
                    false, af, false, bf[j], (short)0, acc[i][j], false, false);
        }
        __syncthreads();
    }

    const int lhalf = (lane >> 4) << 3;
    float* sp = spec + (size_t)b * CCH * TFR;
    for (int i = 0; i < 4; ++i) {
        int mbase = m0 + 64 * wm + 16 * i + lhalf;
        for (int j = 0; j < 2; ++j) {
            int n = 32 * wn + 16 * j + fn;
#pragma unroll
            for (int r = 0; r < 8; ++r) {
                float v = fabsf(acc[i][j][r]);
                size_t o = (size_t)(mbase + r) * TFR + n;
                if (sub) sp[o] -= v; else sp[o] = v;
            }
        }
    }
}

// ---------------------------------------------------------------------------
// Encoder layer: h_out = leaky_relu(W0@h + W1@shift(h,dil) + bias, 0.2) + h
// Dual WMMA GEMM: M=1024 (o), K=1024 (c), N=128 (t), per batch.
// The shift is along t only and the tile spans all t, so the shifted B
// fragment is just column n+dil of the same staged slab (zero beyond t=127).
// ---------------------------------------------------------------------------
__global__ __launch_bounds__(256) void enc_layer_kernel(
    const float* __restrict__ h_in,   // [2][1024][128]
    const half_t* __restrict__ W0,    // [1024][1024] f16
    const half_t* __restrict__ W1,    // [1024][1024] f16
    const float* __restrict__ bias,   // [1024]
    float* __restrict__ h_out,        // [2][1024][128]
    int dil)
{
    const int b  = blockIdx.y;
    const int m0 = blockIdx.x * 128;
    const float* hb = h_in + (size_t)b * CCH * TFR;

    __shared__ half_t As0[128][40];  // [o_local][c_local]
    __shared__ half_t As1[128][40];
    __shared__ half_t Bs[128][40];   // [t][c_local]  (transposed staging)

    const int tid  = threadIdx.x;
    const int lane = tid & 31;
    const int wv   = tid >> 5;
    const int wm   = wv & 1;
    const int wn   = wv >> 1;

    const int fm  = lane & 15;
    const int kbA = (lane < 16) ? 0 : 8;
    const int kbB = (lane < 16) ? 0 : 16;
    const int fn  = lane & 15;

    __builtin_amdgcn_s_wait_tensorcnt(0);  // drain any outstanding TDM traffic

    v8f vzero = {0.f, 0.f, 0.f, 0.f, 0.f, 0.f, 0.f, 0.f};
    v8f acc[4][2];
    for (int i = 0; i < 4; ++i) for (int j = 0; j < 2; ++j) acc[i][j] = vzero;

    const int ar0 = tid >> 2, ack = (tid & 3) * 8;     // A staging task
    const int brp = (tid >> 5) * 2;                    // B staging row pair
    const int bt4 = (tid & 31) * 4;                    // B staging t base

    for (int kk = 0; kk < CCH; kk += 32) {
        // Stage weight tiles: grouped loads, then stores
        const size_t g0 = (size_t)(m0 + ar0) * CCH + kk + ack;
        const size_t g1 = (size_t)(m0 + ar0 + 64) * CCH + kk + ack;
        uint4 w00 = *(const uint4*)(W0 + g0);
        uint4 w01 = *(const uint4*)(W0 + g1);
        uint4 w10 = *(const uint4*)(W1 + g0);
        uint4 w11 = *(const uint4*)(W1 + g1);
        // Stage activation tile: row pairs (r, r+1) and (r+16, r+17)
        float4 v0a = *(const float4*)(hb + (size_t)(kk + brp + 0) * TFR + bt4);
        float4 v0b = *(const float4*)(hb + (size_t)(kk + brp + 1) * TFR + bt4);
        float4 v1a = *(const float4*)(hb + (size_t)(kk + brp + 16) * TFR + bt4);
        float4 v1b = *(const float4*)(hb + (size_t)(kk + brp + 17) * TFR + bt4);

        *(uint4*)&As0[ar0][ack]      = w00;
        *(uint4*)&As0[ar0 + 64][ack] = w01;
        *(uint4*)&As1[ar0][ack]      = w10;
        *(uint4*)&As1[ar0 + 64][ack] = w11;

        {
            float a0[4] = {v0a.x, v0a.y, v0a.z, v0a.w};
            float a1[4] = {v0b.x, v0b.y, v0b.z, v0b.w};
            float c0[4] = {v1a.x, v1a.y, v1a.z, v1a.w};
            float c1[4] = {v1b.x, v1b.y, v1b.z, v1b.w};
#pragma unroll
            for (int j = 0; j < 4; ++j) {
                HP2 p, q;
                p.h[0] = (half_t)a0[j]; p.h[1] = (half_t)a1[j];
                q.h[0] = (half_t)c0[j]; q.h[1] = (half_t)c1[j];
                *(uint*)&Bs[bt4 + j][brp]      = p.u;
                *(uint*)&Bs[bt4 + j][brp + 16] = q.u;
            }
        }
        __syncthreads();

        if (kk + 32 < CCH)  // prefetch next weight tile -> global_prefetch_b8
            __builtin_prefetch(W0 + (size_t)(m0 + (tid >> 1)) * CCH + kk + 32, 0, 1);

        v16h bf0[2], bf1[2];
#pragma unroll
        for (int j = 0; j < 2; ++j) {
            int n  = 32 * wn + 16 * j + fn;
            int ns = n + dil;
            bf0[j] = load_b_frag(&Bs[n][0], kbB);
            v16h t = zero_frag();
            if (ns < TFR) t = load_b_frag(&Bs[ns][0], kbB);
            bf1[j] = t;
        }
#pragma unroll
        for (int i = 0; i < 4; ++i) {
            int m = 64 * wm + 16 * i + fm;
            v16h af0 = load_a_frag(&As0[m][0], kbA);
            v16h af1 = load_a_frag(&As1[m][0], kbA);
#pragma unroll
            for (int j = 0; j < 2; ++j) {
                acc[i][j] = __builtin_amdgcn_wmma_f32_16x16x32_f16(
                    false, af0, false, bf0[j], (short)0, acc[i][j], false, false);
                acc[i][j] = __builtin_amdgcn_wmma_f32_16x16x32_f16(
                    false, af1, false, bf1[j], (short)0, acc[i][j], false, false);
            }
        }
        __syncthreads();
    }

    const int lhalf = (lane >> 4) << 3;
    float* op = h_out + (size_t)b * CCH * TFR;
    for (int i = 0; i < 4; ++i) {
        int mbase = m0 + 64 * wm + 16 * i + lhalf;
        for (int j = 0; j < 2; ++j) {
            int n = 32 * wn + 16 * j + fn;
#pragma unroll
            for (int r = 0; r < 8; ++r) {
                int mm = mbase + r;
                float v = acc[i][j][r] + bias[mm];
                v = (v > 0.f) ? v : 0.2f * v;
                op[(size_t)mm * TFR + n] = v + hb[(size_t)mm * TFR + n];
            }
        }
    }
}

// ---------------------------------------------------------------------------
// Head: attn = relu(w_sw . h[:,t] + b_sw), argmax over t, gather event vector
// ---------------------------------------------------------------------------
__global__ __launch_bounds__(256) void head_kernel(
    const float* __restrict__ h,       // [2][1024][128]
    const float* __restrict__ w_vec,   // [256][1024]
    const float* __restrict__ b_vec,   // [256]
    const float* __restrict__ w_sw,    // [1024]
    const float* __restrict__ b_sw,    // [1]
    float* __restrict__ vecs,          // [2][8][256]
    float* __restrict__ amp,           // [2][8]
    int* __restrict__ idxs,            // [2][8]
    int e)
{
    const int b = blockIdx.x;
    const float* hb = h + (size_t)b * CCH * TFR;
    __shared__ float attn[TFR];
    __shared__ float hcol[CCH];
    __shared__ int sidx;
    const int tid = threadIdx.x;

    if (tid < TFR) {
        float s = b_sw[0];
        for (int c = 0; c < CCH; ++c) s += w_sw[c] * hb[(size_t)c * TFR + tid];
        attn[tid] = fmaxf(s, 0.f);
    }
    __syncthreads();
    if (tid == 0) {
        int bi = 0; float bv = attn[0];
        for (int t = 1; t < TFR; ++t) if (attn[t] > bv) { bv = attn[t]; bi = t; }
        sidx = bi;
        idxs[b * NEV + e] = bi;
        amp[b * NEV + e]  = bv;
    }
    __syncthreads();
    const int p = sidx;
    for (int c = tid; c < CCH; c += 256) hcol[c] = hb[(size_t)c * TFR + p];
    __syncthreads();
    float s = b_vec[tid];
    for (int c = 0; c < CCH; ++c) s += w_vec[(size_t)tid * CCH + c] * hcol[c];
    vecs[((size_t)b * NEV + e) * EMB_D + tid] = s;
}

// ---------------------------------------------------------------------------
// generate() pieces.  slot = b*8 + e0 + (blockIdx.x % n), b = blockIdx.x / n
// ---------------------------------------------------------------------------
__global__ __launch_bounds__(256) void emb_gain_kernel(
    const float* __restrict__ vecs, const float* __restrict__ w_ctx,
    const float* __restrict__ b_ctx, const float* __restrict__ w_verb,
    float* __restrict__ emb, float* __restrict__ gain, int e0, int n)
{
    const int j = blockIdx.x, b = j / n, slot = b * NEV + e0 + (j % n);
    const float* v = vecs + (size_t)slot * EMB_D;
    __shared__ float red[256], red2[256];
    const int tid = threadIdx.x;
    float x = v[tid];
    red[tid]  = x * x;
    red2[tid] = x * w_verb[tid];
    __syncthreads();
    for (int s = 128; s > 0; s >>= 1) {
        if (tid < s) { red[tid] += red[tid + s]; red2[tid] += red2[tid + s]; }
        __syncthreads();
    }
    if (tid == 0) {
        float nr = sqrtf(red[0]) + 1e-8f;
        gain[slot] = 1.f / (1.f + expf(-(red2[0] / nr)));
    }
    float s = b_ctx[tid];
    for (int i = 0; i < EMB_D; ++i) s += v[i] * w_ctx[(size_t)tid * EMB_D + i];
    emb[(size_t)slot * EMB_D + tid] = s;
}

__global__ __launch_bounds__(256) void imprw_kernel(
    const float* __restrict__ emb, const float* __restrict__ w_imp,
    const float* __restrict__ w_res, const float* __restrict__ atom_bias,
    float* __restrict__ imp, float* __restrict__ rw, int e0, int n)
{
    const int j = blockIdx.x, b = j / n, slot = b * NEV + e0 + (j % n);
    __shared__ float es[EMB_D];
    __shared__ float red[256];
    const int tid = threadIdx.x;
    es[tid] = emb[(size_t)slot * EMB_D + tid];
    __syncthreads();

    float* ip = imp + (size_t)slot * NATOM;
    float* rp = rw  + (size_t)slot * NATOM;
    float ssq = 0.f, mx = -1e30f;
    for (int a = tid; a < NATOM; a += 256) {
        const float* wi = w_imp + (size_t)a * EMB_D;
        const float* wr = w_res + (size_t)a * EMB_D;
        float di = 0.f, dr = 0.f;
        for (int i = 0; i < EMB_D; ++i) { float e2 = es[i]; di += e2 * wi[i]; dr += e2 * wr[i]; }
        float ti = tanhf(di);
        ip[a] = ti; ssq += ti * ti;
        dr += atom_bias[a];
        rp[a] = dr; mx = fmaxf(mx, dr);
    }
    red[tid] = ssq; __syncthreads();
    for (int s = 128; s > 0; s >>= 1) { if (tid < s) red[tid] += red[tid + s]; __syncthreads(); }
    const float inorm = 1.f / (sqrtf(red[0]) + 1e-8f);
    __syncthreads();
    red[tid] = mx; __syncthreads();
    for (int s = 128; s > 0; s >>= 1) { if (tid < s) red[tid] = fmaxf(red[tid], red[tid + s]); __syncthreads(); }
    const float gmx = red[0];
    __syncthreads();
    float se = 0.f;
    for (int a = tid; a < NATOM; a += 256) { float ev = expf(rp[a] - gmx); rp[a] = ev; se += ev; }
    red[tid] = se; __syncthreads();
    for (int s = 128; s > 0; s >>= 1) { if (tid < s) red[tid] += red[tid + s]; __syncthreads(); }
    const float inv = 1.f / red[0];
    for (int a = tid; a < NATOM; a += 256) { rp[a] *= inv; ip[a] *= inorm; }
}

__global__ __launch_bounds__(256) void mixed_kernel(
    const float* __restrict__ rw, const float* __restrict__ waves,
    float* __restrict__ mixed, int e0, int n)
{
    const int j = blockIdx.x, b = j / n, slot = b * NEV + e0 + (j % n);
    __shared__ float rs[NATOM];
    const int tid = threadIdx.x;
    for (int a = tid; a < NATOM; a += 256) rs[a] = rw[(size_t)slot * NATOM + a];
    __syncthreads();
    const int s = blockIdx.y * 256 + tid;
    float acc = 0.f;
    for (int a = 0; a < NATOM; ++a) acc += rs[a] * waves[(size_t)a * WAVE_LEN + s];
    mixed[(size_t)slot * WAVE_LEN + s] = acc;
}

// Causal direct convolution: out[t] = sum_{k<=min(t,4095)} imp[k] * mixed[t-k]
__global__ __launch_bounds__(256) void conv_kernel(
    const float* __restrict__ imp, const float* __restrict__ mixed,
    float* __restrict__ convo, int e0, int n)
{
    const int j = blockIdx.x, b = j / n, slot = b * NEV + e0 + (j % n);
    __shared__ float is[NATOM];
    const int tid = threadIdx.x;
    for (int a = tid; a < NATOM; a += 256) is[a] = imp[(size_t)slot * NATOM + a];
    __syncthreads();
    const int t = blockIdx.y * 256 + tid;
    const float* mp = mixed + (size_t)slot * WAVE_LEN;
    const int kmax = (t < NATOM - 1) ? t : (NATOM - 1);
    float acc = 0.f;
    for (int k = 0; k <= kmax; ++k) acc += is[k] * mp[t - k];
    convo[(size_t)slot * WAVE_LEN + t] = acc;
}

__global__ __launch_bounds__(256) void scale_kernel(
    const float* __restrict__ convo, const float* __restrict__ amp,
    const float* __restrict__ gain, float* __restrict__ scale, int e0, int n)
{
    const int j = blockIdx.x, b = j / n, slot = b * NEV + e0 + (j % n);
    __shared__ float red[256];
    const int tid = threadIdx.x;
    const float* cp = convo + (size_t)slot * WAVE_LEN;
    float s = 0.f;
    for (int t = tid; t < WAVE_LEN; t += 256) { float v = cp[t]; s += v * v; }
    red[tid] = s; __syncthreads();
    for (int q = 128; q > 0; q >>= 1) { if (tid < q) red[tid] += red[tid + q]; __syncthreads(); }
    if (tid == 0) {
        float a = amp[slot];
        // unit_norm * amps, then impulse-convolution scales by amp again, then gain
        scale[slot] = a * a * gain[slot] / (sqrtf(red[0]) + 1e-8f);
    }
}

// final = conv with single-impulse "up" => shift by idx*HOP, scale; pad to 32768
__global__ __launch_bounds__(256) void scatter_out_kernel(
    const float* __restrict__ convo, const float* __restrict__ scale,
    const int* __restrict__ idxs, float* __restrict__ dst, int e0, int n, int per_event)
{
    const int j = blockIdx.x, b = j / n, slot = b * NEV + e0 + (j % n);
    const int row = per_event ? b : slot;
    const int t = blockIdx.y * 256 + threadIdx.x;
    const int p = idxs[slot] * HOP;
    const int u = t - p;
    float v = 0.f;
    if (u >= 0 && u < WAVE_LEN) v = convo[(size_t)slot * WAVE_LEN + u] * scale[slot];
    dst[(size_t)row * N_SAMPLES + t] = v;
}

// ---------------------------------------------------------------------------
// Orchestration
// ---------------------------------------------------------------------------
extern "C" void kernel_launch(void* const* d_in, const int* in_sizes, int n_in,
                              void* d_out, int out_size, void* d_ws, size_t ws_size,
                              hipStream_t stream)
{
    const float* x         = (const float*)d_in[0];
    const float* fb        = (const float*)d_in[1];
    const float* enc_w     = (const float*)d_in[2];
    const float* enc_b     = (const float*)d_in[3];
    const float* w_vec     = (const float*)d_in[4];
    const float* b_vec     = (const float*)d_in[5];
    const float* w_sw      = (const float*)d_in[6];
    const float* b_sw      = (const float*)d_in[7];
    const float* w_ctx     = (const float*)d_in[8];
    const float* b_ctx     = (const float*)d_in[9];
    const float* w_imp     = (const float*)d_in[10];
    const float* w_res     = (const float*)d_in[11];
    const float* atom_bias = (const float*)d_in[12];
    const float* waves     = (const float*)d_in[13];
    const float* w_verb    = (const float*)d_in[14];
    (void)in_sizes; (void)n_in; (void)out_size; (void)ws_size;

    char* ws = (char*)d_ws;
    size_t off = 0;
    auto alloc = [&](size_t bytes) -> void* {
        void* p = ws + off;
        off += (bytes + 255) & ~(size_t)255;
        return p;
    };
    half_t* fbh    = (half_t*)alloc((size_t)1024 * 512 * sizeof(half_t));
    half_t* ewh    = (half_t*)alloc((size_t)8 * 2 * 1024 * 1024 * sizeof(half_t));
    float*  spec   = (float*)alloc((size_t)2 * CCH * TFR * sizeof(float));
    float*  hA     = (float*)alloc((size_t)2 * CCH * TFR * sizeof(float));
    float*  hB     = (float*)alloc((size_t)2 * CCH * TFR * sizeof(float));
    float*  vecs   = (float*)alloc((size_t)16 * EMB_D * sizeof(float));
    float*  ampb   = (float*)alloc(16 * sizeof(float));
    float*  gainb  = (float*)alloc(16 * sizeof(float));
    int*    idxb   = (int*)alloc(16 * sizeof(int));
    float*  embb   = (float*)alloc((size_t)16 * EMB_D * sizeof(float));
    float*  impb   = (float*)alloc((size_t)16 * NATOM * sizeof(float));
    float*  rwb    = (float*)alloc((size_t)16 * NATOM * sizeof(float));
    float*  mixb   = (float*)alloc((size_t)16 * WAVE_LEN * sizeof(float));
    float*  convb  = (float*)alloc((size_t)16 * WAVE_LEN * sizeof(float));
    float*  scaleb = (float*)alloc(16 * sizeof(float));
    float*  chb    = (float*)alloc((size_t)2 * N_SAMPLES * sizeof(float));

    // One-time (per launch) weight conversion
    cvt_f16_kernel<<<dim3(256), 256, 0, stream>>>(fb, fbh, 1024 * 512);
    pack_encw_kernel<<<dim3(1024), 256, 0, stream>>>(enc_w, ewh);

    static const int dil[8] = {1, 2, 4, 8, 16, 32, 64, 1};

    // spec = transform(x)
    transform_wmma_kernel<<<dim3(8, 2), 256, 0, stream>>>(x, fbh, spec, 0);

    for (int e = 0; e < NEV; ++e) {
        for (int l = 0; l < 8; ++l) {
            const float* in_p  = (l == 0) ? spec : ((l & 1) ? hA : hB);
            float*       out_p = (l & 1) ? hB : hA;
            const half_t* W0 = ewh + ((size_t)(l * 2 + 0) << 20);
            const half_t* W1 = ewh + ((size_t)(l * 2 + 1) << 20);
            enc_layer_kernel<<<dim3(8, 2), 256, 0, stream>>>(in_p, W0, W1,
                                                             enc_b + l * 1024, out_p, dil[l]);
        }
        head_kernel<<<2, 256, 0, stream>>>(hB, w_vec, b_vec, w_sw, b_sw, vecs, ampb, idxb, e);

        // per-event generate -> ch
        emb_gain_kernel<<<2, 256, 0, stream>>>(vecs, w_ctx, b_ctx, w_verb, embb, gainb, e, 1);
        imprw_kernel<<<2, 256, 0, stream>>>(embb, w_imp, w_res, atom_bias, impb, rwb, e, 1);
        mixed_kernel<<<dim3(2, 32), 256, 0, stream>>>(rwb, waves, mixb, e, 1);
        conv_kernel<<<dim3(2, 32), 256, 0, stream>>>(impb, mixb, convb, e, 1);
        scale_kernel<<<2, 256, 0, stream>>>(convb, ampb, gainb, scaleb, e, 1);
        scatter_out_kernel<<<dim3(2, 128), 256, 0, stream>>>(convb, scaleb, idxb, chb, e, 1, 1);

        // spec -= |transform(ch)|
        transform_wmma_kernel<<<dim3(8, 2), 256, 0, stream>>>(chb, fbh, spec, 1);
    }

    // Final generate over all 16 (b,event) slots -> d_out [2][8][32768]
    emb_gain_kernel<<<16, 256, 0, stream>>>(vecs, w_ctx, b_ctx, w_verb, embb, gainb, 0, 8);
    imprw_kernel<<<16, 256, 0, stream>>>(embb, w_imp, w_res, atom_bias, impb, rwb, 0, 8);
    mixed_kernel<<<dim3(16, 32), 256, 0, stream>>>(rwb, waves, mixb, 0, 8);
    conv_kernel<<<dim3(16, 32), 256, 0, stream>>>(impb, mixb, convb, 0, 8);
    scale_kernel<<<16, 256, 0, stream>>>(convb, ampb, gainb, scaleb, 0, 8);
    scatter_out_kernel<<<dim3(16, 128), 256, 0, stream>>>(convb, scaleb, idxb,
                                                          (float*)d_out, 0, 8, 0);
}
